// CostVolumeLayer_833223655543
// MI455X (gfx1250) — matrix operand order repaired and checked by reading, not compiled, not run on hardware
//
#include <hip/hip_runtime.h>

typedef __attribute__((ext_vector_type(16))) _Float16 v16h;
typedef __attribute__((ext_vector_type(8)))  float    v8f;

#define H_      192
#define W_      384
#define C_      128
#define D_      64
#define REFW    447      // W + 63
#define PITCH   136      // f16 elements per LDS row (128 + 8 pad -> 272B, 16B aligned, bank-spread)
#define WCHUNK  128      // w rows per workgroup (8 waves x 16)
#define REFROWS 192      // ref window rows staged (need 191)
#define NTHREADS 256

__device__ inline unsigned int pk2h(float a, float b) {
  _Float16 ha = (_Float16)a, hb = (_Float16)b;
  unsigned short ua = __builtin_bit_cast(unsigned short, ha);
  unsigned short ub = __builtin_bit_cast(unsigned short, hb);
  return (unsigned int)ua | ((unsigned int)ub << 16);
}

__device__ inline v16h ld_frag(const unsigned short* p0, const unsigned short* p1) {
  union { uint4 u[2]; v16h v; } r;
  r.u[0] = *(const uint4*)p0;   // ds_load_b128
  r.u[1] = *(const uint4*)p1;   // ds_load_b128
  return r.v;
}

// One block: one (b*H+h) pair and one 128-wide w-chunk. 8 waves, each owns a 16-row w-tile
// and accumulates 5 G-tiles (the Gram band) via v_wmma_f32_16x16x32_f16, then scatters the band.
__global__ void __launch_bounds__(NTHREADS)
cost_volume_wmma(const float* __restrict__ refp, const float* __restrict__ auxp,
                 float* __restrict__ out) {
  extern __shared__ unsigned short lds[];   // [WCHUNK+REFROWS][PITCH] f16, aux first then ref

  const int tid   = threadIdx.x;
  const int blk   = blockIdx.x;
  const int chunk = blk % 3;           // W / WCHUNK == 3
  const int bh    = blk / 3;           // b*H + h
  const int w0    = chunk * WCHUNK;

  // ---- Stage aux chunk: rows w0..w0+127, 128 ch, f32 -> f16 (float4 -> 2 packed dwords) ----
  {
    const float* src = auxp + ((size_t)bh * W_ + w0) * C_;
    #pragma unroll
    for (int q = tid; q < WCHUNK * 32; q += NTHREADS) {   // 16 iters, uniform trip count
      int row = q >> 5, qc = q & 31;
      float4 v = *(const float4*)(src + (size_t)row * C_ + qc * 4);
      uint2 p; p.x = pk2h(v.x, v.y); p.y = pk2h(v.z, v.w);
      *(uint2*)(&lds[row * PITCH + qc * 4]) = p;
    }
  }
  // ---- Stage ref window: rows w0..w0+191 (clamped to 446; clamped rows are never used) ----
  {
    const float* src = refp + (size_t)bh * REFW * C_;
    #pragma unroll
    for (int q = tid; q < REFROWS * 32; q += NTHREADS) {  // 24 iters, uniform
      int row = q >> 5, qc = q & 31;
      int gr = w0 + row; gr = (gr > REFW - 1) ? (REFW - 1) : gr;
      float4 v = *(const float4*)(src + (size_t)gr * C_ + qc * 4);
      uint2 p; p.x = pk2h(v.x, v.y); p.y = pk2h(v.z, v.w);
      *(uint2*)(&lds[(WCHUNK + row) * PITCH + qc * 4]) = p;
    }
  }
  __syncthreads();

  // ---- Per-wave WMMA band compute ----
  const int wv   = tid >> 5;       // wave 0..7 -> w-tile
  const int lane = tid & 31;
  const int j    = lane & 15;      // A row (M) / B column (N)
  const int hi   = lane >> 4;      // lane half selects K sub-ranges per ISA layout
  const int wt   = wv * 16;        // w-tile base, relative to w0

  const unsigned short* auxL = lds;
  const unsigned short* refL = lds + WCHUNK * PITCH;

  v8f acc[5];
  #pragma unroll
  for (int t = 0; t < 5; ++t) acc[t] = (v8f){0.f,0.f,0.f,0.f,0.f,0.f,0.f,0.f};

  const int arow = wt + j;
  #pragma unroll
  for (int k0 = 0; k0 < C_; k0 += 32) {
    // A (16x32 f16): lanes 0-15 hold K k0+{0..7,16..23}, lanes 16-31 K k0+{8..15,24..31}
    const int kA = k0 + hi * 8;
    v16h a = ld_frag(auxL + arow * PITCH + kA, auxL + arow * PITCH + kA + 16);
    // B (32x16 f16): lanes 0-15 hold K k0+{0..7,16..23}, lanes 16-31 K k0+{8..15,24..31}, column N = j
    const int kB = k0 + hi * 8;
    #pragma unroll
    for (int t = 0; t < 5; ++t) {
      const unsigned short* bp = refL + (wt + t * 16 + j) * PITCH + kB;
      v16h b = ld_frag(bp, bp + 16);
      acc[t] = __builtin_amdgcn_wmma_f32_16x16x32_f16(
          /*neg_a=*/false, a, /*neg_b=*/false, b,
          /*c_mod=*/(short)0, acc[t], /*reuse_a=*/false, /*reuse_b=*/false);
    }
  }

  // ---- Band extraction: lane holds G[w0+wt+r+8*hi, w0+wt+16t+j]; d = 63 - (16t+j - i) ----
  const size_t outRow = (size_t)bh * W_;
  #pragma unroll
  for (int t = 0; t < 5; ++t) {
    #pragma unroll
    for (int r = 0; r < 8; ++r) {
      int i = r + 8 * hi;
      int s = 16 * t + j - i;
      if (s >= 0 && s < D_) {
        int w = w0 + wt + i;
        out[(outRow + w) * D_ + (D_ - 1 - s)] = acc[t][r];
      }
    }
  }
}

extern "C" void kernel_launch(void* const* d_in, const int* in_sizes, int n_in,
                              void* d_out, int out_size, void* d_ws, size_t ws_size,
                              hipStream_t stream) {
  const float* refp = (const float*)d_in[0];  // [B, H, W+63, C] f32
  const float* auxp = (const float*)d_in[1];  // [B, H, W, C]    f32
  float* out = (float*)d_out;                 // [B, H, W, D]    f32

  const int B = in_sizes[1] / (H_ * W_ * C_);
  const int nblocks = B * H_ * (W_ / WCHUNK);           // chunk fastest -> ref L2 reuse
  const size_t ldsBytes = (size_t)(WCHUNK + REFROWS) * PITCH * sizeof(unsigned short); // 87040

  cost_volume_wmma<<<nblocks, NTHREADS, ldsBytes, stream>>>(refp, auxp, out);
}
